// SAGNNNet_73452530696952
// MI455X (gfx1250) — matrix operand order repaired and checked by reading, not compiled, not run on hardware
//
#include <hip/hip_runtime.h>
#include <hip/hip_bf16.h>

typedef __attribute__((ext_vector_type(16))) _Float16 v16h;
typedef __attribute__((ext_vector_type(8)))  _Float16 v8h;
typedef __attribute__((ext_vector_type(8)))  float    v8f;

#define HID 128
#define NLAB 10
#define NEG_SLOPE 0.2f
#define THREADS 256

static __device__ __forceinline__ float lrelu(float v) { return v > 0.0f ? v : NEG_SLOPE * v; }
// monotone float<->uint mapping so unsigned atomicMax == float max
static __device__ __forceinline__ unsigned fenc(float f) {
    unsigned u = __float_as_uint(f);
    return (u & 0x80000000u) ? ~u : (u | 0x80000000u);
}
static __device__ __forceinline__ float fdec(unsigned e) {
    return __uint_as_float((e & 0x80000000u) ? (e ^ 0x80000000u) : ~e);
}

// ---------------- weight pre-pack into WMMA B-fragment order -----------------
// layout: [(colTile*4 + kstep) * 32 + lane] * 16 contiguous f16
// lane<16 : col = ct*16+lane,    K = ks*32 + 0..7, 16..23
// lane>=16: col = ct*16+lane-16, K = ks*32 + 8..15, 24..31
__global__ void k_pack_w(const float* __restrict__ W, _Float16* __restrict__ Wp,
                         int ncols, int total) {
    int t = blockIdx.x * blockDim.x + threadIdx.x;
    if (t >= total) return;
    int tt   = t & 15;
    int lane = (t >> 4) & 31;
    int ks   = (t >> 9) & 3;
    int ct   = t >> 11;
    int col  = ct * 16 + (lane & 15);
    int K    = ks * 32 + ((tt >> 3) << 4) + ((lane >> 4) << 3) + (tt & 7);
    float v  = (col < ncols) ? W[K * ncols + col] : 0.0f;
    Wp[t] = (_Float16)v;
}

// ---------------- drop mask + masked f16 input activations ------------------
__global__ void k_mask(const float* __restrict__ dh, float* __restrict__ dropb,
                       int N, int R) {
    int i = blockIdx.x * blockDim.x + threadIdx.x;
    if (i >= N) return;
    float m = 0.25f * (dh[i] + dh[N + i] + dh[2 * N + i] + dh[3 * N + i]);
    dropb[i] = (m != 0.0f) ? 1.0f : 0.0f;
    for (int r = 1; r < R; ++r) dropb[r * N + i] = 0.0f;
}

__global__ void k_xh(const float* __restrict__ x, const float* __restrict__ dropb,
                     _Float16* __restrict__ Hh, int N, int ntot) {
    int t = blockIdx.x * blockDim.x + threadIdx.x;
    if (t >= ntot * HID) return;
    int v = t >> 7, f = t & 127;
    float val = (dropb[v] != 0.0f) ? 0.0f : x[(v % N) * HID + f];
    Hh[t] = (_Float16)val;
}

// ---------------- degree / symmetric GCN norm -------------------------------
__global__ void k_fill(float* __restrict__ p, float v, int n) {
    int t = blockIdx.x * blockDim.x + threadIdx.x;
    if (t < n) p[t] = v;
}

__global__ void k_deg_edges(const int* __restrict__ ei, float* __restrict__ deg,
                            int E, int N, int RE) {
    int t = blockIdx.x * blockDim.x + threadIdx.x;
    if (t >= RE) return;
    int r = t / E, e = t - r * E;
    atomicAdd(&deg[r * N + ei[E + e]], 1.0f);
}

__global__ void k_dinv(float* __restrict__ d, int n) {
    int t = blockIdx.x * blockDim.x + threadIdx.x;
    if (t >= n) return;
    float v = d[t];
    d[t] = v > 0.0f ? rsqrtf(v) : 0.0f;
}

// ---------------- WMMA GEMM: C[ntot x (16*NCT)] = A[ntot x 128] * W ----------
// One wave (wave32) per 16-row tile. A fragments (4 k-steps) are loaded once
// into registers and reused across all NCT column tiles -> 8x less A traffic.
template <int LDC, int NCT>
__global__ __launch_bounds__(THREADS)
void k_gemm_wmma(const _Float16* __restrict__ A, const _Float16* __restrict__ Wp,
                 float* __restrict__ C, int ntot) {
    int gid  = (blockIdx.x * blockDim.x + threadIdx.x) >> 5;  // wave id
    int lane = threadIdx.x & 31;
    int rowTiles = (ntot + 15) >> 4;
    if (gid >= rowTiles) return;
    int r0  = gid << 4;
    int row = r0 + (lane & 15);
    if (row >= ntot) row = ntot - 1;
    int khalf = (lane >> 4) << 3;  // 0 or 8 : A-matrix 16-bit lane layout
    const _Float16* arow = A + (size_t)row * HID;

    v16h afrag[4];
#pragma unroll
    for (int ks = 0; ks < 4; ++ks) {
        v8h alo = *(const v8h*)(arow + (ks << 5) + khalf);
        v8h ahi = *(const v8h*)(arow + (ks << 5) + 16 + khalf);
        afrag[ks] = __builtin_shufflevector(alo, ahi, 0, 1, 2, 3, 4, 5, 6, 7,
                                            8, 9, 10, 11, 12, 13, 14, 15);
    }

    // C/D layout: VGPR j -> M = j (lanes 0-15) or j+8 (lanes 16-31), N = lane&15
    int  cr   = r0 + ((lane >> 4) << 3);
    int  cc   = lane & 15;
    bool full = (r0 + 16 <= ntot);

#pragma unroll
    for (int ct = 0; ct < NCT; ++ct) {
        v8f acc = {};
#pragma unroll
        for (int ks = 0; ks < 4; ++ks) {
            v16h b = *(const v16h*)(Wp + ((((ct << 2) + ks) << 5) + lane) * 16);
            acc = __builtin_amdgcn_wmma_f32_16x16x32_f16(
                false, afrag[ks], false, b, (short)0, acc, false, false);
        }
        float* cp = C + (size_t)cr * LDC + ct * 16 + cc;
        if (full) {
#pragma unroll
            for (int j = 0; j < 8; ++j) cp[j * LDC] = acc[j];
        } else {
#pragma unroll
            for (int j = 0; j < 8; ++j)
                if (cr + j < ntot) cp[j * LDC] = acc[j];
        }
    }
}

// ---------------- GCN aggregation -------------------------------------------
__global__ void k_self_norm(const float* __restrict__ hW, const float* __restrict__ dinv,
                            float* __restrict__ agg, int total, int shift) {
    int t = blockIdx.x * blockDim.x + threadIdx.x;
    if (t >= total) return;
    float dv = dinv[t >> shift];
    agg[t] = dv * dv * hW[t];
}

__global__ void k_gcn_edges(const int* __restrict__ ei, const float* __restrict__ dinv,
                            const float* __restrict__ hW, float* __restrict__ agg,
                            int E, int N, int RE) {
    int wid  = (blockIdx.x * blockDim.x + threadIdx.x) >> 5;
    int lane = threadIdx.x & 31;
    if (wid >= RE) return;
    int r = wid / E, e = wid - r * E;
    int s = ei[e] + r * N, d = ei[E + e] + r * N;
    float nrm = dinv[s] * dinv[d];
    const float4 h = *(const float4*)(hW + (size_t)s * HID + lane * 4);
    float* o = agg + (size_t)d * HID + lane * 4;
    atomicAdd(o + 0, nrm * h.x);
    atomicAdd(o + 1, nrm * h.y);
    atomicAdd(o + 2, nrm * h.z);
    atomicAdd(o + 3, nrm * h.w);
}

__global__ void k_bias_act_h16(const float* __restrict__ agg, const float* __restrict__ b,
                               _Float16* __restrict__ Hh, int ntot, int do_relu) {
    int t = blockIdx.x * blockDim.x + threadIdx.x;
    if (t >= ntot * HID) return;
    float v = agg[t] + b[t & 127];
    if (do_relu) v = fmaxf(v, 0.0f);
    Hh[t] = (_Float16)v;
}

// ---------------- GAT -------------------------------------------------------
__global__ void k_alpha(const float* __restrict__ hW, const float* __restrict__ a_s,
                        const float* __restrict__ a_d, float* __restrict__ as_,
                        float* __restrict__ ad_, int ntot) {
    int wid  = (blockIdx.x * blockDim.x + threadIdx.x) >> 5;
    int lane = threadIdx.x & 31;
    if (wid >= ntot) return;
    float4 h  = *(const float4*)(hW + (size_t)wid * HID + lane * 4);
    float4 s4 = *(const float4*)(a_s + lane * 4);
    float4 d4 = *(const float4*)(a_d + lane * 4);
    float ss = h.x * s4.x + h.y * s4.y + h.z * s4.z + h.w * s4.w;
    float dd = h.x * d4.x + h.y * d4.y + h.z * d4.z + h.w * d4.w;
    for (int o = 16; o > 0; o >>= 1) {
        ss += __shfl_xor(ss, o, 32);
        dd += __shfl_xor(dd, o, 32);
    }
    if (lane == 0) { as_[wid] = ss; ad_[wid] = dd; }
}

__global__ void k_max_init(const float* __restrict__ as_, const float* __restrict__ ad_,
                           unsigned* __restrict__ mmax, int ntot) {
    int t = blockIdx.x * blockDim.x + threadIdx.x;
    if (t >= ntot) return;
    mmax[t] = fenc(lrelu(as_[t] + ad_[t]));  // self-loop term
}

__global__ void k_max_edges(const int* __restrict__ ei, const float* __restrict__ as_,
                            const float* __restrict__ ad_, unsigned* __restrict__ mmax,
                            int E, int N, int RE) {
    int t = blockIdx.x * blockDim.x + threadIdx.x;
    if (t >= RE) return;
    int r = t / E, e = t - r * E;
    int s = ei[e] + r * N, d = ei[E + e] + r * N;
    atomicMax(&mmax[d], fenc(lrelu(as_[s] + ad_[d])));
}

__global__ void k_gat_self(const float* __restrict__ as_, const float* __restrict__ ad_,
                           const unsigned* __restrict__ mmax, const float* __restrict__ hW,
                           float* __restrict__ denom, float* __restrict__ agg, int ntot) {
    int wid  = (blockIdx.x * blockDim.x + threadIdx.x) >> 5;
    int lane = threadIdx.x & 31;
    if (wid >= ntot) return;
    float es = lrelu(as_[wid] + ad_[wid]);
    float w  = __expf(es - fdec(mmax[wid]));
    if (lane == 0) denom[wid] = w;
    float4 h = *(const float4*)(hW + (size_t)wid * HID + lane * 4);
    float4* o = (float4*)(agg + (size_t)wid * HID + lane * 4);
    *o = make_float4(w * h.x, w * h.y, w * h.z, w * h.w);
}

__global__ void k_gat_edges(const int* __restrict__ ei, const float* __restrict__ as_,
                            const float* __restrict__ ad_, const unsigned* __restrict__ mmax,
                            const float* __restrict__ hW, float* __restrict__ denom,
                            float* __restrict__ agg, int E, int N, int RE) {
    int wid  = (blockIdx.x * blockDim.x + threadIdx.x) >> 5;
    int lane = threadIdx.x & 31;
    if (wid >= RE) return;
    int r = wid / E, e = wid - r * E;
    int s = ei[e] + r * N, d = ei[E + e] + r * N;
    float w = __expf(lrelu(as_[s] + ad_[d]) - fdec(mmax[d]));
    if (lane == 0) atomicAdd(&denom[d], w);
    float4 h = *(const float4*)(hW + (size_t)s * HID + lane * 4);
    float* o = agg + (size_t)d * HID + lane * 4;
    atomicAdd(o + 0, w * h.x);
    atomicAdd(o + 1, w * h.y);
    atomicAdd(o + 2, w * h.z);
    atomicAdd(o + 3, w * h.w);
}

__global__ void k_gat_fin(const float* __restrict__ agg, const float* __restrict__ denom,
                          const float* __restrict__ b, _Float16* __restrict__ Hh, int ntot) {
    int t = blockIdx.x * blockDim.x + threadIdx.x;
    if (t >= ntot * HID) return;
    float v = agg[t] / denom[t >> 7] + b[t & 127];
    Hh[t] = (_Float16)fmaxf(v, 0.0f);
}

// ---------------- final 16-wide GCN + log_softmax ---------------------------
__global__ void k_gcn_edges16(const int* __restrict__ ei, const float* __restrict__ dinv,
                              const float* __restrict__ hW2, float* __restrict__ agg2,
                              int E, int N, int total) {
    int t = blockIdx.x * blockDim.x + threadIdx.x;
    if (t >= total) return;
    int c = t & 15, we = t >> 4;
    int r = we / E, e = we - r * E;
    int s = ei[e] + r * N, d = ei[E + e] + r * N;
    float nrm = dinv[s] * dinv[d];
    atomicAdd(&agg2[d * 16 + c], nrm * hW2[s * 16 + c]);
}

__global__ void k_logsoftmax(const float* __restrict__ agg2, const float* __restrict__ b2,
                             float* __restrict__ out, int ntot) {
    int v = blockIdx.x * blockDim.x + threadIdx.x;
    if (v >= ntot) return;
    float l[NLAB], mx = -3.0e38f;
    for (int c = 0; c < NLAB; ++c) { l[c] = agg2[v * 16 + c] + b2[c]; mx = fmaxf(mx, l[c]); }
    float s = 0.0f;
    for (int c = 0; c < NLAB; ++c) s += __expf(l[c] - mx);
    float lse = mx + __logf(s);
    for (int c = 0; c < NLAB; ++c) out[v * NLAB + c] = l[c] - lse;
}

// ---------------- host orchestration ----------------------------------------
static inline int ngrid(long n) { return (int)((n + THREADS - 1) / THREADS); }

extern "C" void kernel_launch(void* const* d_in, const int* in_sizes, int n_in,
                              void* d_out, int out_size, void* d_ws, size_t ws_size,
                              hipStream_t stream) {
    const float* x     = (const float*)d_in[0];
    const int*   ei    = (const int*)d_in[1];
    const float* dhalf = (const float*)d_in[2];
    const float* W1    = (const float*)d_in[3];
    const float* b1    = (const float*)d_in[4];
    const float* Wg    = (const float*)d_in[5];
    const float* a_src = (const float*)d_in[6];
    const float* a_dst = (const float*)d_in[7];
    const float* bg    = (const float*)d_in[8];
    const float* W2    = (const float*)d_in[9];
    const float* b2    = (const float*)d_in[10];

    const int N    = in_sizes[2] / 4;            // drop_half is [4,1,N]
    const int E    = in_sizes[1] / 2;            // edge_index is [2,E]
    const int R    = out_size / (11 * N);        // out = R*N bool + R*N*10 logp
    const int ntot = R * N;
    const int RE   = R * E;
    const int rowTiles = (ntot + 15) / 16;

    // workspace carve-up (256B aligned)
    char* base = (char*)d_ws;
    size_t off = 0;
    auto take = [&](size_t bytes) -> void* {
        void* p = base + off;
        off += (bytes + 255) & ~(size_t)255;
        return p;
    };
    _Float16* Hh   = (_Float16*)take((size_t)ntot * HID * sizeof(_Float16));
    float*    hW   = (float*)take((size_t)ntot * HID * sizeof(float));
    float*    agg  = (float*)take((size_t)ntot * HID * sizeof(float));
    float*    dinv = (float*)take((size_t)ntot * sizeof(float));
    float*    as_  = (float*)take((size_t)ntot * sizeof(float));
    float*    ad_  = (float*)take((size_t)ntot * sizeof(float));
    unsigned* mmax = (unsigned*)take((size_t)ntot * sizeof(unsigned));
    float*    den  = (float*)take((size_t)ntot * sizeof(float));
    _Float16* Wp1  = (_Float16*)take(8 * 2048 * sizeof(_Float16));
    _Float16* Wpg  = (_Float16*)take(8 * 2048 * sizeof(_Float16));
    _Float16* Wp2  = (_Float16*)take(1 * 2048 * sizeof(_Float16));
    (void)ws_size; (void)n_in;

    float* dropb = (float*)d_out;        // [R*N]
    float* logp  = (float*)d_out + ntot; // [R*N*10]

    // weight packing (B-fragment order, f16)
    k_pack_w<<<ngrid(8 * 2048), THREADS, 0, stream>>>(W1, Wp1, HID, 8 * 2048);
    k_pack_w<<<ngrid(8 * 2048), THREADS, 0, stream>>>(Wg, Wpg, HID, 8 * 2048);
    k_pack_w<<<ngrid(1 * 2048), THREADS, 0, stream>>>(W2, Wp2, NLAB, 1 * 2048);

    // drop mask + masked f16 input
    k_mask<<<ngrid(N), THREADS, 0, stream>>>(dhalf, dropb, N, R);
    k_xh<<<ngrid((long)ntot * HID), THREADS, 0, stream>>>(x, dropb, Hh, N, ntot);

    // degrees -> D^{-1/2} (self-loops give the initial 1)
    k_fill<<<ngrid(ntot), THREADS, 0, stream>>>(dinv, 1.0f, ntot);
    k_deg_edges<<<ngrid(RE), THREADS, 0, stream>>>(ei, dinv, E, N, RE);
    k_dinv<<<ngrid(ntot), THREADS, 0, stream>>>(dinv, ntot);

    // ---- GCN layer 1: h = relu(norm-agg(x@W1) + b1) ----
    k_gemm_wmma<HID, 8><<<ngrid((long)rowTiles * 32), THREADS, 0, stream>>>(
        Hh, Wp1, hW, ntot);
    k_self_norm<<<ngrid((long)ntot * HID), THREADS, 0, stream>>>(hW, dinv, agg,
                                                                 ntot * HID, 7);
    k_gcn_edges<<<ngrid((long)RE * 32), THREADS, 0, stream>>>(ei, dinv, hW, agg, E, N, RE);
    k_bias_act_h16<<<ngrid((long)ntot * HID), THREADS, 0, stream>>>(agg, b1, Hh, ntot, 1);

    // ---- 2x GAT layers ----
    for (int layer = 0; layer < 2; ++layer) {
        k_gemm_wmma<HID, 8><<<ngrid((long)rowTiles * 32), THREADS, 0, stream>>>(
            Hh, Wpg, hW, ntot);
        k_alpha<<<ngrid((long)ntot * 32), THREADS, 0, stream>>>(hW, a_src, a_dst,
                                                                as_, ad_, ntot);
        k_max_init<<<ngrid(ntot), THREADS, 0, stream>>>(as_, ad_, mmax, ntot);
        k_max_edges<<<ngrid(RE), THREADS, 0, stream>>>(ei, as_, ad_, mmax, E, N, RE);
        k_gat_self<<<ngrid((long)ntot * 32), THREADS, 0, stream>>>(as_, ad_, mmax, hW,
                                                                   den, agg, ntot);
        k_gat_edges<<<ngrid((long)RE * 32), THREADS, 0, stream>>>(ei, as_, ad_, mmax, hW,
                                                                  den, agg, E, N, RE);
        k_gat_fin<<<ngrid((long)ntot * HID), THREADS, 0, stream>>>(agg, den, bg, Hh, ntot);
    }

    // ---- GCN layer 2 (128 -> 10, padded to 16 cols) + log_softmax ----
    k_gemm_wmma<16, 1><<<ngrid((long)rowTiles * 32), THREADS, 0, stream>>>(
        Hh, Wp2, hW, ntot);
    k_self_norm<<<ngrid((long)ntot * 16), THREADS, 0, stream>>>(hW, dinv, agg,
                                                                ntot * 16, 4);
    k_gcn_edges16<<<ngrid((long)RE * 16), THREADS, 0, stream>>>(ei, dinv, hW, agg,
                                                                E, N, RE * 16);
    k_logsoftmax<<<ngrid(ntot), THREADS, 0, stream>>>(agg, b2, logp, ntot);
}
